// conv_1d_49752901157365
// MI455X (gfx1250) — matrix operand order repaired and checked by reading, not compiled, not run on hardware
//
#include <hip/hip_runtime.h>

// ---------------------------------------------------------------------------
// Conv1D (valid cross-correlation) as implicit GEMM on CDNA5 WMMA (gfx1250).
//   out[n,i,o] = sum_{t,c} w[o,t,c] * x[n,i+t,c] + b[o]
// GEMM view: M = B*Lout (row (n,i) of A = contiguous 1280 floats of x),
//            N = K,     (col o of B = contiguous 1280 floats of w[o])
//            red = KS*C = 1280.
// Precision: bf16x3 split (hi/lo) + f32 accumulation via
//            V_WMMA_F32_16X16X32_BF16  (acc += Ah*Bh + Ah*Bl + Al*Bh).
// ---------------------------------------------------------------------------

#define B_    32
#define L_    2048
#define C_    256
#define K_    512
#define KS_   5
#define LOUT  (L_ - KS_ + 1)     // 2044
#define RED   (KS_ * C_)         // 1280
#define MTOT  (B_ * LOUT)        // 65408  (multiple of 64)
#define XN    (B_ * L_ * C_)     // 16777216 elements
#define WN    (K_ * RED)         // 655360 elements

typedef __bf16        v16bf __attribute__((ext_vector_type(16)));
typedef float         v8f   __attribute__((ext_vector_type(8)));
typedef unsigned int  v4u   __attribute__((ext_vector_type(4)));

union Frag { v4u q[2]; v16bf v; };   // 32 bytes = 8 VGPRs = one bf16 A/B fragment

__device__ __forceinline__ unsigned short f32_to_bf16_rne(float f) {
  unsigned int u = __float_as_uint(f);
  unsigned int r = u + 0x7FFFu + ((u >> 16) & 1u);   // round-to-nearest-even
  return (unsigned short)(r >> 16);
}
__device__ __forceinline__ float bf16_to_f32(unsigned short h) {
  return __uint_as_float(((unsigned int)h) << 16);
}

// ---------------------------------------------------------------------------
// Prep: split fp32 x and w into bf16 hi/lo pairs (one element per thread).
// ---------------------------------------------------------------------------
__global__ void conv1d_prep_bf16x2(const float* __restrict__ x,
                                   const float* __restrict__ w,
                                   unsigned short* __restrict__ xh,
                                   unsigned short* __restrict__ xl,
                                   unsigned short* __restrict__ wh,
                                   unsigned short* __restrict__ wl) {
  unsigned int idx = blockIdx.x * 256u + threadIdx.x;
  if (idx < (unsigned int)XN) {
    float f = x[idx];
    unsigned short h = f32_to_bf16_rne(f);
    xh[idx] = h;
    xl[idx] = f32_to_bf16_rne(f - bf16_to_f32(h));
  } else if (idx < (unsigned int)(XN + WN)) {
    unsigned int j = idx - (unsigned int)XN;
    float f = w[j];
    unsigned short h = f32_to_bf16_rne(f);
    wh[j] = h;
    wl[j] = f32_to_bf16_rne(f - bf16_to_f32(h));
  }
}

// ---------------------------------------------------------------------------
// GEMM: 256 threads = 8 waves as 2(M) x 4(N); each wave computes 32x32 of out
// with 4 f32 accumulators. 40 k-steps of 32; 12 WMMAs per step per wave.
// x (67MB) + w (2.6MB) are L2-resident, so fragments stream from global in
// native WMMA register layout (contiguous per-lane runs; no shuffles).
// ---------------------------------------------------------------------------
__global__ __launch_bounds__(256)
void conv1d_wmma_bf16x3(const unsigned short* __restrict__ xh,
                        const unsigned short* __restrict__ xl,
                        const unsigned short* __restrict__ wh,
                        const unsigned short* __restrict__ wl,
                        const float* __restrict__ bias,
                        float* __restrict__ out) {
  const int lane = threadIdx.x & 31;
  const int wave = threadIdx.x >> 5;
  const int wm   = wave & 1;        // 0..1 : M direction
  const int wn   = wave >> 1;       // 0..3 : N direction
  const int sel  = lane >> 4;       // half-wave select for fragment layout
  const int l16  = lane & 15;

  const int Mbase = blockIdx.y * 64 + wm * 32;
  const int Nbase = blockIdx.x * 128 + wn * 32;

  // A-row element offsets (two 16-row fragments). Row m -> (n,i); the 1280
  // reduction elements are contiguous starting at x[(n*L + i)*C].
  int rowA[2];
#pragma unroll
  for (int a = 0; a < 2; ++a) {
    int m = Mbase + a * 16 + l16;
    int n = m / LOUT;
    int i = m - n * LOUT;
    rowA[a] = (n * L_ + i) * C_ + sel * 8;     // lane<16: K 0..7/16..23 ; lane>=16: +8
  }
  // B-column element offsets (two 16-col fragments). Column o is w[o][0..1279].
  int colB[2];
#pragma unroll
  for (int b = 0; b < 2; ++b) {
    int o = Nbase + b * 16 + l16;
    colB[b] = o * RED + sel * 8;               // lane<16: K 0..7/16..23 ; lane>=16: +8
  }

  v8f acc[2][2];
  v8f zero = {};
#pragma unroll
  for (int a = 0; a < 2; ++a)
#pragma unroll
    for (int b = 0; b < 2; ++b) acc[a][b] = zero;

  for (int rb = 0; rb < RED; rb += 32) {
    Frag Ah[2], Al[2], Bh[2], Bl[2];
#pragma unroll
    for (int a = 0; a < 2; ++a) {
      const unsigned short* ph = xh + rowA[a] + rb;
      const unsigned short* pl = xl + rowA[a] + rb;
      Ah[a].q[0] = *(const v4u*)(ph);          // K run 0..7   (this half-wave)
      Ah[a].q[1] = *(const v4u*)(ph + 16);     // K run 16..23 (this half-wave)
      Al[a].q[0] = *(const v4u*)(pl);
      Al[a].q[1] = *(const v4u*)(pl + 16);
    }
#pragma unroll
    for (int b = 0; b < 2; ++b) {
      const unsigned short* ph = wh + colB[b] + rb;
      const unsigned short* pl = wl + colB[b] + rb;
      Bh[b].q[0] = *(const v4u*)(ph);          // K run 0..7   (this half-wave)
      Bh[b].q[1] = *(const v4u*)(ph + 16);     // K run 16..23 (this half-wave)
      Bl[b].q[0] = *(const v4u*)(pl);
      Bl[b].q[1] = *(const v4u*)(pl + 16);
    }
#pragma unroll
    for (int a = 0; a < 2; ++a)
#pragma unroll
      for (int b = 0; b < 2; ++b) {
        acc[a][b] = __builtin_amdgcn_wmma_f32_16x16x32_bf16(
            false, Ah[a].v, false, Bh[b].v, (short)0, acc[a][b], false, false);
        acc[a][b] = __builtin_amdgcn_wmma_f32_16x16x32_bf16(
            false, Ah[a].v, false, Bl[b].v, (short)0, acc[a][b], false, false);
        acc[a][b] = __builtin_amdgcn_wmma_f32_16x16x32_bf16(
            false, Al[a].v, false, Bh[b].v, (short)0, acc[a][b], false, false);
      }
  }

  // Epilogue: C/D layout -> element (VGPR j, lane l) = (M = j + 8*sel, N = l16).
#pragma unroll
  for (int b = 0; b < 2; ++b) {
    const int col = Nbase + b * 16 + l16;
    const float bv = bias[col];
#pragma unroll
    for (int a = 0; a < 2; ++a) {
      const int row0 = Mbase + a * 16 + sel * 8;
      float* po = out + (size_t)row0 * K_ + col;
#pragma unroll
      for (int j = 0; j < 8; ++j) {
        po[(size_t)j * K_] = acc[a][b][j] + bv;
      }
    }
  }
}

// ---------------------------------------------------------------------------
extern "C" void kernel_launch(void* const* d_in, const int* in_sizes, int n_in,
                              void* d_out, int out_size, void* d_ws, size_t ws_size,
                              hipStream_t stream) {
  const float* x = (const float*)d_in[0];   // [B, L, C]
  const float* w = (const float*)d_in[1];   // [K, KS, C] == [K][RED]
  const float* b = (const float*)d_in[2];   // [K]
  float* out = (float*)d_out;               // [B, Lout, K] == [MTOT][K]

  unsigned short* xh = (unsigned short*)d_ws;
  unsigned short* xl = xh + (size_t)XN;
  unsigned short* wh = xl + (size_t)XN;
  unsigned short* wl = wh + (size_t)WN;     // total ws use: 2*(XN+WN)*2 B ~ 66.5 MiB

  const unsigned int total = XN + WN;
  conv1d_prep_bf16x2<<<(total + 255u) / 256u, 256, 0, stream>>>(x, w, xh, xl, wh, wl);

  dim3 grid(K_ / 128, MTOT / 64);           // (4, 1022)
  conv1d_wmma_bf16x3<<<grid, 256, 0, stream>>>(xh, xl, wh, wl, b, out);
}
